// MessagePassing_30477087932672
// MI455X (gfx1250) — matrix operand order repaired and checked by reading, not compiled, not run on hardware
//
#include <hip/hip_runtime.h>
#include <hip/hip_bf16.h>

typedef __attribute__((ext_vector_type(16))) _Float16 v16h;
typedef __attribute__((ext_vector_type(8)))  _Float16 v8h;
typedef __attribute__((ext_vector_type(8)))  float    v8f;

#define C_    256
#define H_    96
#define W_    96
#define HW_   (H_*W_)
#define HP_   102          // 96 + 3 top + 3 bottom
#define WP_   112          // 96 + 3 left + 13 right (16-aligned, covers K up to 31)
#define HWP_  (HP_*WP_)
#define TAPS  49
#define OUT_HW (384*384)

// ---------------------------------------------------------------------------
// 1) Row-normalize per-pixel window weights: nw[s,p] = w[s,p] / (sum_s + eps)
// ---------------------------------------------------------------------------
__global__ __launch_bounds__(256) void norm_weights(const float* __restrict__ wgt,
                                                    float* __restrict__ nw) {
    int p = blockIdx.x * 256 + threadIdx.x;
    if (p >= HW_) return;
    float s = 0.f;
    #pragma unroll
    for (int i = 0; i < TAPS; ++i) s += wgt[i * HW_ + p];
    float inv = 1.f / (s + 1e-5f);
    #pragma unroll
    for (int i = 0; i < TAPS; ++i) nw[i * HW_ + p] = wgt[i * HW_ + p] * inv;
}

// ---------------------------------------------------------------------------
// 2) Build zero-padded f16 image xpad0 from f32 input; zero borders of xpad1.
//    Runs every call => deterministic across graph replays.
// ---------------------------------------------------------------------------
__global__ __launch_bounds__(256) void pad_init(const float* __restrict__ x,
                                                _Float16* __restrict__ xpad0,
                                                _Float16* __restrict__ xpad1) {
    int i = blockIdx.x * 256 + threadIdx.x;
    if (i >= C_ * HWP_) return;
    int ch  = i / HWP_;
    int rem = i - ch * HWP_;
    int pr  = rem / WP_;
    int pc  = rem - pr * WP_;
    bool interior = (pr >= 3) && (pr < 3 + H_) && (pc >= 3) && (pc < 3 + W_);
    _Float16 v = (_Float16)0.f;
    if (interior) v = (_Float16)x[ch * HW_ + (pr - 3) * W_ + (pc - 3)];
    xpad0[i] = v;
    if (!interior) xpad1[i] = (_Float16)0.f;   // mp_step writes the interior
}

// ---------------------------------------------------------------------------
// 3) One message-passing step via V_WMMA_F32_16X16X32_F16, branch-free feeds.
//    Block = 256 threads = 8 waves; covers (row r, 16-col tile) x 128 channels
//    (blockIdx.z picks channel half). A: 2x global_load_b128/dy from padded
//    image. B: 2x ds_load_b128/dy from precomputed banded tiles in LDS.
// ---------------------------------------------------------------------------
template <bool OUT_F32>
__global__ __launch_bounds__(256) void mp_step(const _Float16* __restrict__ xin,
                                               const float* __restrict__ nw,
                                               void* __restrict__ xout_) {
    __shared__ float snw[TAPS * 16];       // 49 x 16 per-pixel weights
    __shared__ v8h   Bl8[7 * 16 * 4];      // B tiles: [dy][n][K(32)] as 4x v8h

    const int c0   = blockIdx.x * 16;      // pixel-column tile base
    const int r    = blockIdx.y;           // pixel row
    const int tid  = threadIdx.x;
    const int lane = tid & 31;
    const int wave = tid >> 5;
    const int ch0  = (blockIdx.z * 8 + wave) * 16;

    // --- stage this tile's normalized weights (coalesced)
    for (int i = tid; i < TAPS * 16; i += 256) {
        int s = i >> 4, n = i & 15;
        snw[i] = nw[s * HW_ + r * W_ + c0 + n];
    }
    __syncthreads();

    // --- build banded B tiles branch-free: Bl[dy][n][K] = nw[dy*7+(K-n)] if
    //     0 <= K-n <= 6 else 0 (zeroes also mask unused K >= 22 of A)
    _Float16* Blh = (_Float16*)Bl8;
    #pragma unroll
    for (int it = 0; it < 14; ++it) {
        int i  = tid + it * 256;           // 7*16*32 = 3584 elements
        int K  = i & 31;
        int n  = (i >> 5) & 15;
        int dy = i >> 9;
        int dx = K - n;
        int dxc = min(max(dx, 0), 6);
        float v = snw[(dy * 7 + dxc) * 16 + n];
        bool ok = (dx >= 0) && (dx <= 6);
        Blh[i] = ok ? (_Float16)v : (_Float16)0.f;
    }
    __syncthreads();

    const int  nIdx   = lane & 15;
    const bool hiHalf = lane >= 16;

    // padded row index of output row r, tap dy: (r + dy - 3) + 3 = r + dy
    // padded col index of K:                   (c0 - 3 + K) + 3 = c0 + K
    const _Float16* abase = xin + (size_t)(ch0 + nIdx) * HWP_
                                + (size_t)r * WP_ + c0 + (hiHalf ? 8 : 0);
    const int bbase = (nIdx << 2) + (hiHalf ? 1 : 0);   // v8h units within a dy

    v8f acc = {0.f, 0.f, 0.f, 0.f, 0.f, 0.f, 0.f, 0.f};

    #pragma unroll
    for (int dy = 0; dy < 7; ++dy) {
        const _Float16* ap = abase + (size_t)dy * WP_;
        v8h a0 = *(const v8h*)(ap);        // halves 0..7
        v8h a1 = *(const v8h*)(ap + 16);   // halves 8..15 (K + 16)
        v16h a = __builtin_shufflevector(a0, a1,
                 0,1,2,3,4,5,6,7,8,9,10,11,12,13,14,15);

        v8h b0 = Bl8[dy * 64 + bbase];
        v8h b1 = Bl8[dy * 64 + bbase + 2];
        v16h b = __builtin_shufflevector(b0, b1,
                 0,1,2,3,4,5,6,7,8,9,10,11,12,13,14,15);

        acc = __builtin_amdgcn_wmma_f32_16x16x32_f16(
            /*neg_a=*/false, a, /*neg_b=*/false, b,
            /*c_mod=*/(short)0, acc, /*reuse_a=*/false, /*reuse_b=*/false);
    }

    // --- D writeback: VGPR v <-> M = v (+8 upper lanes), N = nIdx
    #pragma unroll
    for (int v = 0; v < 8; ++v) {
        int m = hiHalf ? v + 8 : v;
        if (OUT_F32) {
            ((float*)xout_)[(size_t)(ch0 + m) * HW_ + r * W_ + c0 + nIdx] = acc[v];
        } else {
            ((_Float16*)xout_)[(size_t)(ch0 + m) * HWP_ + (r + 3) * WP_ + (c0 + nIdx + 3)]
                = (_Float16)acc[v];
        }
    }
}

// ---------------------------------------------------------------------------
// 4) 1x1 conv 256 -> 3 channels (+bias). One thread per pixel; coalesced x.
// ---------------------------------------------------------------------------
__global__ __launch_bounds__(256) void conv1x1(const float* __restrict__ x,
                                               const float* __restrict__ cw,
                                               const float* __restrict__ cb,
                                               float* __restrict__ y) {
    int p = blockIdx.x * 256 + threadIdx.x;
    if (p >= HW_) return;
    float a0 = cb[0], a1 = cb[1], a2 = cb[2];
    for (int c = 0; c < C_; ++c) {
        float xv = x[c * HW_ + p];
        a0 = fmaf(xv, cw[c], a0);
        a1 = fmaf(xv, cw[C_ + c], a1);
        a2 = fmaf(xv, cw[2 * C_ + c], a2);
    }
    y[p]           = a0;
    y[HW_ + p]     = a1;
    y[2 * HW_ + p] = a2;
}

// ---------------------------------------------------------------------------
// 5) Bilinear upsample 96x96 -> 384x384, half-pixel centers, edge clamp.
// ---------------------------------------------------------------------------
__global__ __launch_bounds__(256) void resize_bilinear(const float* __restrict__ y,
                                                       float* __restrict__ out) {
    int i = blockIdx.x * 256 + threadIdx.x;
    if (i >= 3 * OUT_HW) return;
    int o   = i / OUT_HW;
    int rem = i - o * OUT_HW;
    int oy  = rem / 384;
    int ox  = rem - oy * 384;

    float sy = (oy + 0.5f) * 0.25f - 0.5f;
    float sx = (ox + 0.5f) * 0.25f - 0.5f;
    int y0 = (int)floorf(sy), x0 = (int)floorf(sx);
    float fy = sy - (float)y0, fx = sx - (float)x0;
    int y0c = min(max(y0, 0), H_ - 1);
    int y1c = min(max(y0 + 1, 0), H_ - 1);
    int x0c = min(max(x0, 0), W_ - 1);
    int x1c = min(max(x0 + 1, 0), W_ - 1);

    const float* yp = y + o * HW_;
    float v00 = yp[y0c * W_ + x0c], v01 = yp[y0c * W_ + x1c];
    float v10 = yp[y1c * W_ + x0c], v11 = yp[y1c * W_ + x1c];
    float top = v00 + fx * (v01 - v00);
    float bot = v10 + fx * (v11 - v10);
    out[i] = top + fy * (bot - top);
}

// ---------------------------------------------------------------------------
static inline size_t align_up(size_t v, size_t a) { return (v + a - 1) & ~(a - 1); }

extern "C" void kernel_launch(void* const* d_in, const int* in_sizes, int n_in,
                              void* d_out, int out_size, void* d_ws, size_t ws_size,
                              hipStream_t stream) {
    const float* input  = (const float*)d_in[0];  // (1,256,96,96)
    const float* weight = (const float*)d_in[1];  // (1,49,9216)
    const float* conv_w = (const float*)d_in[2];  // (3,256)
    const float* conv_b = (const float*)d_in[3];  // (3,)
    float* out = (float*)d_out;                   // (1,3,384,384)

    // Carve workspace
    char* ws = (char*)d_ws;
    size_t off = 0;
    float*    nw    = (float*)(ws + off);    off = align_up(off + (size_t)TAPS * HW_ * 4, 256);
    _Float16* xpad0 = (_Float16*)(ws + off); off = align_up(off + (size_t)C_ * HWP_ * 2, 256);
    _Float16* xpad1 = (_Float16*)(ws + off); off = align_up(off + (size_t)C_ * HWP_ * 2, 256);
    float*    x2    = (float*)(ws + off);    off = align_up(off + (size_t)C_ * HW_ * 4, 256);
    float*    y3    = (float*)(ws + off);    off = align_up(off + (size_t)3 * HW_ * 4, 256);
    (void)ws_size; (void)in_sizes; (void)n_in; (void)out_size;

    // 1) normalize weights
    norm_weights<<<(HW_ + 255) / 256, 256, 0, stream>>>(weight, nw);

    // 2) padded f16 image (+ zero xpad1 borders)
    int npad = C_ * HWP_;
    pad_init<<<(npad + 255) / 256, 256, 0, stream>>>(input, xpad0, xpad1);

    // 3) two WMMA message-passing steps
    dim3 grid(W_ / 16, H_, 2);
    mp_step<false><<<grid, 256, 0, stream>>>(xpad0, nw, (void*)xpad1);
    mp_step<true ><<<grid, 256, 0, stream>>>(xpad1, nw, (void*)x2);

    // 4) 1x1 conv to 3 channels
    conv1x1<<<(HW_ + 255) / 256, 256, 0, stream>>>(x2, conv_w, conv_b, y3);

    // 5) bilinear upsample to 384x384
    resize_bilinear<<<(3 * OUT_HW + 255) / 256, 256, 0, stream>>>(y3, out);
}
